// IDLMM_nn_21414706938039
// MI455X (gfx1250) — matrix operand order repaired and checked by reading, not compiled, not run on hardware
//
#include <hip/hip_runtime.h>
#include <hip/hip_bf16.h>

// ---------------------------------------------------------------------------
// Model dims (from reference)
// ---------------------------------------------------------------------------
#define BB      2048
#define NGENE   3008
#define SS      44
#define DMM     32
#define JJHN    (SS*DMM)          // 1408
#define NDRUG   2048
#define GSS     16
#define HH      6
#define N_LEAF  64
#define N_MID   16
#define N_TERMS (N_LEAF + N_MID + 1)   // 81
#define NHEADC  4
#define HDC     8
#define DFFC    2048
#define MSB     (SS*BB)           // 90112
#define EPSF    1e-5f

typedef __attribute__((ext_vector_type(16))) __bf16 v16bf;
typedef __attribute__((ext_vector_type(8)))  __bf16 v8bf;
typedef __attribute__((ext_vector_type(8)))  float  v8f;

#define GLOBAL_AS __attribute__((address_space(1)))
#define LDS_AS    __attribute__((address_space(3)))

#if defined(__AMDGCN__) && __has_builtin(__builtin_amdgcn_global_load_async_to_lds_b32)
#define HAVE_ASYNC_LDS 1
#else
#define HAVE_ASYNC_LDS 0
#endif

// ---------------------------------------------------------------------------
// helpers
// ---------------------------------------------------------------------------
__device__ __forceinline__ __bf16 f2bf(float f) {
    union { float f; unsigned u; } un; un.f = f;
    unsigned short h = (unsigned short)((un.u + 0x8000u) >> 16);
    __bf16 r; __builtin_memcpy(&r, &h, 2);
    return r;
}

// async (or fallback) 4-byte global -> LDS stage
__device__ __forceinline__ void stage_b32(const __bf16* src, __bf16* ldsDst) {
#if HAVE_ASYNC_LDS
    __builtin_amdgcn_global_load_async_to_lds_b32(
        (GLOBAL_AS int*)(void*)const_cast<__bf16*>(src),
        (LDS_AS int*)(void*)ldsDst, 0, 0);
#else
    *(unsigned*)ldsDst = *(const unsigned*)src;
#endif
}

__device__ __forceinline__ void stage_wait() {
#if HAVE_ASYNC_LDS
#if __has_builtin(__builtin_amdgcn_s_wait_asynccnt)
    __builtin_amdgcn_s_wait_asynccnt(0);
#else
    asm volatile("s_wait_asynccnt 0x0" ::: "memory");
#endif
#endif
}

// A-fragment: 16x32 bf16 tile at (m0,k0) of row-major A (leading dim lda).
// Caller guarantees m0+16 <= M, k0+32 <= K, 16B-aligned rows.
__device__ __forceinline__ v16bf load_a_frag(const __bf16* A, int lda,
                                             int m0, int k0) {
    int lane  = threadIdx.x & 31;
    int m     = m0 + (lane & 15);
    int khalf = (lane >> 4) * 16;
    const __bf16* p = A + (size_t)m * lda + k0 + khalf;
    v8bf lo = *(const v8bf*)(p);
    v8bf hi = *(const v8bf*)(p + 8);
    return __builtin_shufflevector(lo, hi, 0,1,2,3,4,5,6,7,8,9,10,11,12,13,14,15);
}

// B-fragment (32x16) read out of an LDS-staged 16-row x 32-col tile
// (row = n within tile, 64 B per row). lane n = lane&15, k = (lane>>4)*16 + i.
__device__ __forceinline__ v16bf load_b_frag_lds(const __bf16* Bs) {
    int lane = threadIdx.x & 31;
    const __bf16* p = Bs + (lane & 15) * 32 + ((lane >> 4) << 4);
    v8bf lo = *(const v8bf*)(p);          // ds_load_b128
    v8bf hi = *(const v8bf*)(p + 8);      // ds_load_b128
    return __builtin_shufflevector(lo, hi, 0,1,2,3,4,5,6,7,8,9,10,11,12,13,14,15);
}

// B-fragment straight from global W (N,K) row-major (used by ffn_fused).
__device__ __forceinline__ v16bf load_b_fragT(const __bf16* W, int ldw,
                                              int n0, int k0) {
    int lane  = threadIdx.x & 31;
    int n     = n0 + (lane & 15);
    int kbase = k0 + ((lane >> 4) << 4);
    const __bf16* p = W + (size_t)n * ldw + kbase;
    v8bf lo = *(const v8bf*)(p);
    v8bf hi = *(const v8bf*)(p + 8);
    return __builtin_shufflevector(lo, hi, 0,1,2,3,4,5,6,7,8,9,10,11,12,13,14,15);
}

// ---------------------------------------------------------------------------
// Generic batched WMMA GEMM:  C[t] = act(A[t] (MxK) * W[t]^T (KxN) + bias[t])
// 8 waves / 256-thread block; each wave owns one 16x16 output tile; all waves
// share n0, so the B tile is staged once per K-step into LDS via async
// global->LDS loads (1 KB), then each wave reads it as two ds_load_b128.
// Requires M % 128 == 0, K % 32 == 0 (true for all launches below).
// grid = (ceil(N/16), M/(16*8), T)
// ---------------------------------------------------------------------------
template<bool NGUARD>
__global__ __launch_bounds__(256)
void wmma_gemm_bf16(const __bf16* __restrict__ A, const __bf16* __restrict__ W,
                    const float* __restrict__ bias, float* __restrict__ C,
                    int M, int N, int K,
                    long long aBatch, long long wBatch, long long bBatch,
                    long long cBatch, int act) {
    __shared__ __align__(16) __bf16 Bs[16 * 32];
    int wave = threadIdx.x >> 5;
    int lane = threadIdx.x & 31;
    int n0 = blockIdx.x * 16;
    int m0 = (blockIdx.y * 8 + wave) * 16;
    int t  = blockIdx.z;
    A += (size_t)t * aBatch;
    W += (size_t)t * wBatch;
    C += (size_t)t * cBatch;
    if (bias) bias += (size_t)t * bBatch;

    // staging role: thread i stages 4 bytes of row (i>>4), elems 2*(i&15)
    int srow = threadIdx.x >> 4;
    int scol = (threadIdx.x & 15) * 2;
    const __bf16* wsrc = W + (size_t)(n0 + srow) * K + scol;
    bool rowOK = !NGUARD || (n0 + srow < N);

    v8f acc = {0.f,0.f,0.f,0.f,0.f,0.f,0.f,0.f};
    for (int k0 = 0; k0 < K; k0 += 32) {
        __syncthreads();                       // prior-step Bs reads done
        if (rowOK) stage_b32(wsrc + k0, &Bs[srow * 32 + scol]);
        else       *(unsigned*)&Bs[srow * 32 + scol] = 0u;
        stage_wait();
        __syncthreads();                       // staged tile visible
        if (k0 + 32 < K)
            __builtin_prefetch(&A[(size_t)(m0 + (threadIdx.x & 15)) * K + k0 + 32], 0, 1);
        v16bf a = load_a_frag(A, K, m0, k0);
        v16bf b = load_b_frag_lds(Bs);
        acc = __builtin_amdgcn_wmma_f32_16x16x32_bf16(false, a, false, b,
                                                      (short)0, acc, false, false);
    }
    int n  = n0 + (lane & 15);
    int mb = m0 + ((lane >> 4) << 3);
    if (!NGUARD || n < N) {
        float bv = bias ? bias[n] : 0.0f;
#pragma unroll
        for (int r = 0; r < 8; ++r) {
            float v = acc[r] + bv;
            if (act == 1)      v = tanhf(v);
            else if (act == 2) v = fmaxf(v, 0.0f);
            C[(size_t)(mb + r) * N + n] = v;
        }
    }
}

// ---------------------------------------------------------------------------
// Fused FFN: Out = X + relu(X*W1^T + b1)*W2^T + b2   (DM=32, DFF=2048)
// 4 waves / 128-thread block; each wave owns a 16-row tile of X and a private
// 16x32 bf16 LDS slice used to turn the lin1 C-fragment into the lin2
// A-fragment. 4 WMMA per 32-wide DFF step. Avoids the 738 MB intermediate.
// ---------------------------------------------------------------------------
__global__ __launch_bounds__(128)
void ffn_fused(const __bf16* __restrict__ Xbf, const float* __restrict__ Xf,
               const __bf16* __restrict__ W1, const float* __restrict__ b1,
               const __bf16* __restrict__ W2, const float* __restrict__ b2,
               float* __restrict__ Out, int M) {
    __shared__ __align__(16) __bf16 hsh[4][16 * 32];
    int wave = threadIdx.x >> 5;
    int lane = threadIdx.x & 31;
    int m0   = (blockIdx.x * 4 + wave) * 16;
    __bf16* hs = hsh[wave];

    v16bf a1 = load_a_frag(Xbf, DMM, m0, 0);
    v8f c2a = {0.f,0.f,0.f,0.f,0.f,0.f,0.f,0.f};
    v8f c2b = {0.f,0.f,0.f,0.f,0.f,0.f,0.f,0.f};

    for (int kk = 0; kk < DFFC / 32; ++kk) {
#pragma unroll
        for (int half = 0; half < 2; ++half) {
            int n1 = kk * 32 + half * 16;
            v16bf bfrag = load_b_fragT(W1, DMM, n1, 0);
            v8f h = {0.f,0.f,0.f,0.f,0.f,0.f,0.f,0.f};
            h = __builtin_amdgcn_wmma_f32_16x16x32_bf16(false, a1, false, bfrag,
                                                        (short)0, h, false, false);
            int ncol = lane & 15;
            float bb = b1[n1 + ncol];
            int mrow = ((lane >> 4) << 3);
#pragma unroll
            for (int r = 0; r < 8; ++r) {
                float v = fmaxf(h[r] + bb, 0.0f);
                hs[(mrow + r) * 32 + half * 16 + ncol] = f2bf(v);
            }
        }
        __syncthreads();
        v16bf a2;
        {
            int m     = lane & 15;
            int khalf = (lane >> 4) * 16;
            const __bf16* p = hs + m * 32 + khalf;
            v8bf lo = *(const v8bf*)(p);          // ds_load_b128
            v8bf hi = *(const v8bf*)(p + 8);      // ds_load_b128
            a2 = __builtin_shufflevector(lo, hi, 0,1,2,3,4,5,6,7,8,9,10,11,12,13,14,15);
        }
        __syncthreads();
        int kg = kk * 32;
        v16bf b2lo = load_b_fragT(W2, DFFC, 0,  kg);
        v16bf b2hi = load_b_fragT(W2, DFFC, 16, kg);
        c2a = __builtin_amdgcn_wmma_f32_16x16x32_bf16(false, a2, false, b2lo,
                                                      (short)0, c2a, false, false);
        c2b = __builtin_amdgcn_wmma_f32_16x16x32_bf16(false, a2, false, b2hi,
                                                      (short)0, c2b, false, false);
    }

    int l15 = lane & 15;
    int mb  = m0 + ((lane >> 4) << 3);
#pragma unroll
    for (int r = 0; r < 8; ++r) {
        size_t row = (size_t)(mb + r) * DMM;
        Out[row + l15]      = Xf[row + l15]      + c2a[r] + b2[l15];
        Out[row + 16 + l15] = Xf[row + 16 + l15] + c2b[r] + b2[16 + l15];
    }
}

// ---------------------------------------------------------------------------
// f32 -> bf16 (strided rows)
// ---------------------------------------------------------------------------
__global__ void cvt_bf16_k(const float* __restrict__ in, __bf16* __restrict__ out,
                           long long rows, int cols, long long inStride) {
    long long idx = (long long)blockIdx.x * blockDim.x + threadIdx.x;
    if (idx >= rows * (long long)cols) return;
    long long r = idx / cols;
    int c = (int)(idx % cols);
    out[r * cols + c] = f2bf(in[r * inStride + c]);
}

// src0 (S,B,DM) f32 + bf16 from x slice
__global__ void build_src_k(const float* __restrict__ x, float* __restrict__ f,
                            __bf16* __restrict__ bf) {
    long long idx = (long long)blockIdx.x * blockDim.x + threadIdx.x;
    if (idx >= (long long)SS * BB * DMM) return;
    int c = (int)(idx % DMM);
    int b = (int)((idx / DMM) % BB);
    int s = (int)(idx / ((long long)DMM * BB));
    float v = x[(long long)b * (NGENE + JJHN + NDRUG) + NGENE + s * DMM + c];
    f[idx]  = v;
    bf[idx] = f2bf(v);
}

// ---------------------------------------------------------------------------
// BN over rows: X (T,M,C), group per (t,c), population variance.
// ---------------------------------------------------------------------------
__global__ __launch_bounds__(256)
void bn_k(const float* __restrict__ X, const float* __restrict__ g,
          const float* __restrict__ b, float* __restrict__ Y, int M, int C) {
    int grp = blockIdx.x;
    int t = grp / C, c = grp % C;
    const float* x = X + (size_t)t * M * C + c;
    float*       y = Y + (size_t)t * M * C + c;
    __shared__ float s1[256], s2[256];
    float sum = 0.f, sq = 0.f;
    for (int m = threadIdx.x; m < M; m += blockDim.x) {
        float v = x[(size_t)m * C];
        sum += v; sq += v * v;
    }
    s1[threadIdx.x] = sum; s2[threadIdx.x] = sq;
    __syncthreads();
    for (int s = 128; s > 0; s >>= 1) {
        if ((int)threadIdx.x < s) {
            s1[threadIdx.x] += s1[threadIdx.x + s];
            s2[threadIdx.x] += s2[threadIdx.x + s];
        }
        __syncthreads();
    }
    float mean = s1[0] / M;
    float var  = s2[0] / M - mean * mean;
    float inv  = rsqrtf(var + EPSF);
    float gg = g[grp], bb = b[grp];
    for (int m = threadIdx.x; m < M; m += blockDim.x)
        y[(size_t)m * C] = gg * (x[(size_t)m * C] - mean) * inv + bb;
}

// ---------------------------------------------------------------------------
// Small batched linear: C[t] = act(A[t](MxK) * W[t]^T + bias[t])  (VALU)
// ---------------------------------------------------------------------------
__global__ void small_lin_k(const float* __restrict__ A, const float* __restrict__ W,
                            const float* __restrict__ bias, float* __restrict__ C,
                            int T, int M, int N, int K,
                            long long aB, long long wB, long long bB, long long cB,
                            int act) {
    long long idx = (long long)blockIdx.x * blockDim.x + threadIdx.x;
    long long total = (long long)T * M * N;
    if (idx >= total) return;
    int n = (int)(idx % N);
    long long rem = idx / N;
    int m = (int)(rem % M);
    int t = (int)(rem / M);
    const float* a = A + t * aB + (long long)m * K;
    const float* w = W + t * wB + (long long)n * K;
    float acc = bias ? bias[t * bB + n] : 0.f;
    for (int k = 0; k < K; ++k) acc += a[k] * w[k];
    if (act == 1) acc = tanhf(acc);
    C[t * cB + (long long)m * N + n] = acc;
}

// ---------------------------------------------------------------------------
// tree glue
// ---------------------------------------------------------------------------
__global__ void build_mid_in_k(const float* __restrict__ leaf_out,
                               const float* __restrict__ g_out,
                               float* __restrict__ mid_in) {
    long long idx = (long long)blockIdx.x * blockDim.x + threadIdx.x;
    if (idx >= (long long)N_MID * BB * 40) return;
    int c = (int)(idx % 40);
    int b = (int)((idx / 40) % BB);
    int t = (int)(idx / (40LL * BB));
    float v;
    if (c < 4 * HH) {
        int j = c / HH, ch = c % HH;
        v = leaf_out[((size_t)(t * 4 + j) * BB + b) * HH + ch];
    } else {
        v = g_out[((size_t)(N_LEAF + t) * BB + b) * GSS + (c - 4 * HH)];
    }
    mid_in[idx] = v;
}

__global__ void build_root_in_k(const float* __restrict__ mid_out,
                                const float* __restrict__ g_out,
                                float* __restrict__ root_in) {
    long long idx = (long long)blockIdx.x * blockDim.x + threadIdx.x;
    if (idx >= (long long)BB * 112) return;
    int c = (int)(idx % 112);
    int b = (int)(idx / 112);
    float v;
    if (c < N_MID * HH) {
        int t = c / HH, ch = c % HH;
        v = mid_out[((size_t)t * BB + b) * HH + ch];
    } else {
        v = g_out[((size_t)(N_TERMS - 1) * BB + b) * GSS + (c - N_MID * HH)];
    }
    root_in[idx] = v;
}

__global__ void concat2_k(const float* __restrict__ a, const float* __restrict__ b,
                          float* __restrict__ out, int C1, int C2) {
    long long idx = (long long)blockIdx.x * blockDim.x + threadIdx.x;
    int C = C1 + C2;
    if (idx >= (long long)BB * C) return;
    int c = (int)(idx % C);
    int r = (int)(idx / C);
    out[idx] = (c < C1) ? a[(long long)r * C1 + c] : b[(long long)r * C2 + (c - C1)];
}

// ---------------------------------------------------------------------------
// attention: per (b,h); S=44, HD=8; two-pass softmax, K/V staged in LDS
// ---------------------------------------------------------------------------
__global__ __launch_bounds__(64)
void attn_k(const float* __restrict__ qkv, float* __restrict__ o) {
    int b = blockIdx.x, h = blockIdx.y;
    __shared__ float Ks[SS][HDC], Vs[SS][HDC];
    int tid = threadIdx.x;
    for (int i = tid; i < SS * HDC; i += blockDim.x) {
        int s = i >> 3, d = i & 7;
        long long base = ((long long)s * BB + b) * (3 * DMM) + h * HDC + d;
        Ks[s][d] = qkv[base + DMM];
        Vs[s][d] = qkv[base + 2 * DMM];
    }
    __syncthreads();
    if (tid < SS) {
        int s = tid;
        float q[HDC];
        long long qb = ((long long)s * BB + b) * (3 * DMM) + h * HDC;
#pragma unroll
        for (int d = 0; d < HDC; ++d) q[d] = qkv[qb + d] * 0.35355339059327373f;
        float mx = -1e30f;
        for (int k = 0; k < SS; ++k) {
            float dot = 0.f;
#pragma unroll
            for (int d = 0; d < HDC; ++d) dot += q[d] * Ks[k][d];
            mx = fmaxf(mx, dot);
        }
        float sum = 0.f;
        float acc[HDC] = {0.f,0.f,0.f,0.f,0.f,0.f,0.f,0.f};
        for (int k = 0; k < SS; ++k) {
            float dot = 0.f;
#pragma unroll
            for (int d = 0; d < HDC; ++d) dot += q[d] * Ks[k][d];
            float e = __expf(dot - mx);
            sum += e;
#pragma unroll
            for (int d = 0; d < HDC; ++d) acc[d] += e * Vs[k][d];
        }
        long long ob = ((long long)s * BB + b) * DMM + h * HDC;
#pragma unroll
        for (int d = 0; d < HDC; ++d) o[ob + d] = acc[d] / sum;
    }
}

// residual + layernorm over last dim 32; optional bf16 mirror
__global__ void res_ln_k(const float* __restrict__ a, const float* __restrict__ delta,
                         const float* __restrict__ g, const float* __restrict__ b,
                         float* __restrict__ outf, __bf16* __restrict__ outbf,
                         long long rows) {
    long long r = (long long)blockIdx.x * blockDim.x + threadIdx.x;
    if (r >= rows) return;
    float v[DMM];
    float mean = 0.f;
#pragma unroll
    for (int c = 0; c < DMM; ++c) {
        float x = a[r * DMM + c] + (delta ? delta[r * DMM + c] : 0.f);
        v[c] = x; mean += x;
    }
    mean *= (1.0f / DMM);
    float var = 0.f;
#pragma unroll
    for (int c = 0; c < DMM; ++c) { float d = v[c] - mean; var += d * d; }
    var *= (1.0f / DMM);
    float inv = rsqrtf(var + EPSF);
#pragma unroll
    for (int c = 0; c < DMM; ++c) {
        float y = g[c] * (v[c] - mean) * inv + b[c];
        outf[r * DMM + c] = y;
        if (outbf) outbf[r * DMM + c] = f2bf(y);
    }
}

__global__ void mean_s_k(const float* __restrict__ src, float* __restrict__ out) {
    long long idx = (long long)blockIdx.x * blockDim.x + threadIdx.x;
    if (idx >= (long long)BB * DMM) return;
    int c = (int)(idx % DMM);
    int b = (int)(idx / DMM);
    float s = 0.f;
    for (int si = 0; si < SS; ++si) s += src[((long long)si * BB + b) * DMM + c];
    out[idx] = s * (1.0f / SS);
}

__global__ void aux_final_k(const float* __restrict__ f1, const float* __restrict__ f2,
                            const float* f1aW, const float* f1ab,
                            const float* f1oW, const float* f1ob,
                            const float* f2aW, const float* f2ab,
                            const float* f2oW, const float* f2ob,
                            float* __restrict__ out) {
    int b = blockIdx.x * blockDim.x + threadIdx.x;
    if (b >= BB) return;
    float t1 = f1ab[0];
#pragma unroll
    for (int k = 0; k < HH; ++k) t1 += f1[(long long)b * HH + k] * f1aW[k];
    t1 = tanhf(t1);
    float a1 = t1 * f1oW[0] + f1ob[0];
    float t2 = f2ab[0];
#pragma unroll
    for (int k = 0; k < HH; ++k) t2 += f2[(long long)b * HH + k] * f2aW[k];
    t2 = tanhf(t2);
    float a2 = t2 * f2oW[0] + f2ob[0];
    out[b] = 0.5f * (a1 + a2);
}

// ---------------------------------------------------------------------------
// host orchestration
// ---------------------------------------------------------------------------
static inline unsigned cdiv(long long a, long long b) { return (unsigned)((a + b - 1) / b); }

extern "C" void kernel_launch(void* const* d_in, const int* in_sizes, int n_in,
                              void* d_out, int out_size, void* d_ws, size_t ws_size,
                              hipStream_t stream) {
    (void)in_sizes; (void)n_in; (void)out_size; (void)ws_size;
    const float* x        = (const float*)d_in[0];
    const float* gene_W   = (const float*)d_in[1];
    const float* gene_b   = (const float*)d_in[2];
    const float* leaf_W   = (const float*)d_in[3];
    const float* leaf_b   = (const float*)d_in[4];
    const float* leaf_g   = (const float*)d_in[5];
    const float* leaf_be  = (const float*)d_in[6];
    const float* mid_W    = (const float*)d_in[7];
    const float* mid_b    = (const float*)d_in[8];
    const float* mid_g    = (const float*)d_in[9];
    const float* mid_be   = (const float*)d_in[10];
    const float* root_W   = (const float*)d_in[11];
    const float* root_b   = (const float*)d_in[12];
    const float* root_g   = (const float*)d_in[13];
    const float* root_be  = (const float*)d_in[14];
    const float* Wqkv     = (const float*)d_in[15];
    const float* bqkv     = (const float*)d_in[16];
    const float* Wo       = (const float*)d_in[17];
    const float* bo       = (const float*)d_in[18];
    const float* ln1_g    = (const float*)d_in[19];
    const float* ln1_b    = (const float*)d_in[20];
    const float* ln2_g    = (const float*)d_in[21];
    const float* ln2_b    = (const float*)d_in[22];
    const float* lin1_W   = (const float*)d_in[23];
    const float* lin1_b   = (const float*)d_in[24];
    const float* lin2_W   = (const float*)d_in[25];
    const float* lin2_b   = (const float*)d_in[26];
    const float* jjh_W    = (const float*)d_in[27];
    const float* jjh_b    = (const float*)d_in[28];
    const float* jjh_g    = (const float*)d_in[29];
    const float* jjh_be   = (const float*)d_in[30];
    const float* d0W = (const float*)d_in[31];
    const float* d0b = (const float*)d_in[32];
    const float* d0g = (const float*)d_in[33];
    const float* d0be= (const float*)d_in[34];
    const float* d1W = (const float*)d_in[35];
    const float* d1b = (const float*)d_in[36];
    const float* d1g = (const float*)d_in[37];
    const float* d1be= (const float*)d_in[38];
    const float* d2W = (const float*)d_in[39];
    const float* d2b = (const float*)d_in[40];
    const float* d2g = (const float*)d_in[41];
    const float* d2be= (const float*)d_in[42];
    const float* f1_W = (const float*)d_in[43];
    const float* f1_b = (const float*)d_in[44];
    const float* f1_g = (const float*)d_in[45];
    const float* f1_be= (const float*)d_in[46];
    const float* f1aW = (const float*)d_in[47];
    const float* f1ab = (const float*)d_in[48];
    const float* f1oW = (const float*)d_in[49];
    const float* f1ob = (const float*)d_in[50];
    const float* f2_W = (const float*)d_in[51];
    const float* f2_b = (const float*)d_in[52];
    const float* f2_g = (const float*)d_in[53];
    const float* f2_be= (const float*)d_in[54];
    const float* f2aW = (const float*)d_in[55];
    const float* f2ab = (const float*)d_in[56];
    const float* f2oW = (const float*)d_in[57];
    const float* f2ob = (const float*)d_in[58];

    // workspace bump allocator (256-B aligned -> all vector loads 16-B aligned)
    size_t off = 0;
    auto alloc = [&](size_t bytes) -> void* {
        void* p = (char*)d_ws + off;
        off += (bytes + 255) & ~(size_t)255;
        return p;
    };
    const int XSTR = NGENE + JJHN + NDRUG;

    __bf16* gene_bf   = (__bf16*)alloc((size_t)BB * NGENE * 2);
    __bf16* geneW_bf  = (__bf16*)alloc((size_t)N_TERMS * GSS * NGENE * 2);
    float*  g_out     = (float*) alloc((size_t)N_TERMS * BB * GSS * 4);
    float*  leaf_h    = (float*) alloc((size_t)N_LEAF * BB * HH * 4);
    float*  leaf_out  = (float*) alloc((size_t)N_LEAF * BB * HH * 4);
    float*  mid_in    = (float*) alloc((size_t)N_MID * BB * 40 * 4);
    float*  mid_h     = (float*) alloc((size_t)N_MID * BB * HH * 4);
    float*  mid_out   = (float*) alloc((size_t)N_MID * BB * HH * 4);
    float*  root_in   = (float*) alloc((size_t)BB * 112 * 4);
    float*  root_h    = (float*) alloc((size_t)BB * HH * 4);
    float*  root_out  = (float*) alloc((size_t)BB * HH * 4);
    float*  src0_f    = (float*) alloc((size_t)MSB * DMM * 4);
    __bf16* src0_bf   = (__bf16*)alloc((size_t)MSB * DMM * 2);
    __bf16* Wqkv_bf   = (__bf16*)alloc((size_t)3 * DMM * DMM * 2);
    __bf16* Wo_bf     = (__bf16*)alloc((size_t)DMM * DMM * 2);
    __bf16* lin1W_bf  = (__bf16*)alloc((size_t)DFFC * DMM * 2);
    __bf16* lin2W_bf  = (__bf16*)alloc((size_t)DMM * DFFC * 2);
    __bf16* d0W_bf    = (__bf16*)alloc((size_t)100 * NDRUG * 2);
    float*  qkv       = (float*) alloc((size_t)MSB * 3 * DMM * 4);
    float*  attn_o    = (float*) alloc((size_t)MSB * DMM * 4);
    __bf16* ao_bf     = (__bf16*)alloc((size_t)MSB * DMM * 2);
    float*  proj      = (float*) alloc((size_t)MSB * DMM * 4);
    float*  src1_f    = (float*) alloc((size_t)MSB * DMM * 4);
    __bf16* src1_bf   = (__bf16*)alloc((size_t)MSB * DMM * 2);
    float*  ffres     = (float*) alloc((size_t)MSB * DMM * 4);
    float*  src2_f    = (float*) alloc((size_t)MSB * DMM * 4);
    float*  jjh_mean  = (float*) alloc((size_t)BB * DMM * 4);
    float*  jjh_h     = (float*) alloc((size_t)BB * HH * 4);
    float*  jjh_out   = (float*) alloc((size_t)BB * HH * 4);
    __bf16* drug_bf   = (__bf16*)alloc((size_t)BB * NDRUG * 2);
    float*  d0_h      = (float*) alloc((size_t)BB * 100 * 4);
    float*  d0o       = (float*) alloc((size_t)BB * 100 * 4);
    float*  d1_h      = (float*) alloc((size_t)BB * 50 * 4);
    float*  d1o      = (float*) alloc((size_t)BB * 50 * 4);
    float*  d2_h      = (float*) alloc((size_t)BB * HH * 4);
    float*  d2o       = (float*) alloc((size_t)BB * HH * 4);
    float*  f1_in     = (float*) alloc((size_t)BB * 2 * HH * 4);
    float*  f1_h      = (float*) alloc((size_t)BB * HH * 4);
    float*  f1o       = (float*) alloc((size_t)BB * HH * 4);
    float*  f2_in     = (float*) alloc((size_t)BB * 2 * HH * 4);
    float*  f2_h      = (float*) alloc((size_t)BB * HH * 4);
    float*  f2o       = (float*) alloc((size_t)BB * HH * 4);

    // ---- conversions ----
    cvt_bf16_k<<<cdiv((long long)BB * NGENE, 256), 256, 0, stream>>>(
        x, gene_bf, BB, NGENE, XSTR);
    cvt_bf16_k<<<cdiv((long long)N_TERMS * GSS * NGENE, 256), 256, 0, stream>>>(
        gene_W, geneW_bf, (long long)N_TERMS * GSS, NGENE, NGENE);
    cvt_bf16_k<<<cdiv(3LL * DMM * DMM, 256), 256, 0, stream>>>(Wqkv, Wqkv_bf, 3 * DMM, DMM, DMM);
    cvt_bf16_k<<<cdiv((long long)DMM * DMM, 256), 256, 0, stream>>>(Wo, Wo_bf, DMM, DMM, DMM);
    cvt_bf16_k<<<cdiv((long long)DFFC * DMM, 256), 256, 0, stream>>>(lin1_W, lin1W_bf, DFFC, DMM, DMM);
    cvt_bf16_k<<<cdiv((long long)DMM * DFFC, 256), 256, 0, stream>>>(lin2_W, lin2W_bf, DMM, DFFC, DFFC);
    cvt_bf16_k<<<cdiv(100LL * NDRUG, 256), 256, 0, stream>>>(d0W, d0W_bf, 100, NDRUG, NDRUG);
    cvt_bf16_k<<<cdiv((long long)BB * NDRUG, 256), 256, 0, stream>>>(
        x + NGENE + JJHN, drug_bf, BB, NDRUG, XSTR);

    // ---- gene einsum: 81-term batched WMMA GEMM (N=16 exact -> no guard) ----
    wmma_gemm_bf16<false><<<dim3(1, BB / (16 * 8), N_TERMS), 256, 0, stream>>>(
        gene_bf, geneW_bf, gene_b, g_out, BB, GSS, NGENE,
        0LL, (long long)GSS * NGENE, GSS, (long long)BB * GSS, 0);

    // ---- ontology tree (tiny VALU) ----
    small_lin_k<<<cdiv((long long)N_LEAF * BB * HH, 256), 256, 0, stream>>>(
        g_out, leaf_W, leaf_b, leaf_h, N_LEAF, BB, HH, GSS,
        (long long)BB * GSS, (long long)HH * GSS, HH, (long long)BB * HH, 1);
    bn_k<<<N_LEAF * HH, 256, 0, stream>>>(leaf_h, leaf_g, leaf_be, leaf_out, BB, HH);
    build_mid_in_k<<<cdiv((long long)N_MID * BB * 40, 256), 256, 0, stream>>>(leaf_out, g_out, mid_in);
    small_lin_k<<<cdiv((long long)N_MID * BB * HH, 256), 256, 0, stream>>>(
        mid_in, mid_W, mid_b, mid_h, N_MID, BB, HH, 40,
        (long long)BB * 40, 40LL * HH, HH, (long long)BB * HH, 1);
    bn_k<<<N_MID * HH, 256, 0, stream>>>(mid_h, mid_g, mid_be, mid_out, BB, HH);
    build_root_in_k<<<cdiv((long long)BB * 112, 256), 256, 0, stream>>>(mid_out, g_out, root_in);
    small_lin_k<<<cdiv((long long)BB * HH, 256), 256, 0, stream>>>(
        root_in, root_W, root_b, root_h, 1, BB, HH, 112, 0, 0, 0, 0, 1);
    bn_k<<<HH, 256, 0, stream>>>(root_h, root_g, root_be, root_out, BB, HH);

    // ---- transformer ----
    build_src_k<<<cdiv((long long)SS * BB * DMM, 256), 256, 0, stream>>>(x, src0_f, src0_bf);
    wmma_gemm_bf16<false><<<dim3(cdiv(3 * DMM, 16), MSB / (16 * 8), 1), 256, 0, stream>>>(
        src0_bf, Wqkv_bf, bqkv, qkv, MSB, 3 * DMM, DMM, 0, 0, 0, 0, 0);
    attn_k<<<dim3(BB, NHEADC), 64, 0, stream>>>(qkv, attn_o);
    cvt_bf16_k<<<cdiv((long long)MSB * DMM, 256), 256, 0, stream>>>(attn_o, ao_bf, MSB, DMM, DMM);
    wmma_gemm_bf16<false><<<dim3(DMM / 16, MSB / (16 * 8), 1), 256, 0, stream>>>(
        ao_bf, Wo_bf, bo, proj, MSB, DMM, DMM, 0, 0, 0, 0, 0);
    res_ln_k<<<cdiv(MSB, 256), 256, 0, stream>>>(src0_f, proj, ln1_g, ln1_b, src1_f, src1_bf, MSB);
    ffn_fused<<<MSB / (16 * 4), 128, 0, stream>>>(src1_bf, src1_f, lin1W_bf, lin1_b,
                                                  lin2W_bf, lin2_b, ffres, MSB);
    res_ln_k<<<cdiv(MSB, 256), 256, 0, stream>>>(ffres, nullptr, ln2_g, ln2_b,
                                                 src2_f, (__bf16*)nullptr, MSB);
    mean_s_k<<<cdiv((long long)BB * DMM, 256), 256, 0, stream>>>(src2_f, jjh_mean);
    small_lin_k<<<cdiv((long long)BB * HH, 256), 256, 0, stream>>>(
        jjh_mean, jjh_W, jjh_b, jjh_h, 1, BB, HH, DMM, 0, 0, 0, 0, 1);
    bn_k<<<HH, 256, 0, stream>>>(jjh_h, jjh_g, jjh_be, jjh_out, BB, HH);

    // ---- drug MLP (N=100 ragged -> guarded variant) ----
    wmma_gemm_bf16<true><<<dim3(cdiv(100, 16), BB / (16 * 8), 1), 256, 0, stream>>>(
        drug_bf, d0W_bf, d0b, d0_h, BB, 100, NDRUG, 0, 0, 0, 0, 1);
    bn_k<<<100, 256, 0, stream>>>(d0_h, d0g, d0be, d0o, BB, 100);
    small_lin_k<<<cdiv((long long)BB * 50, 256), 256, 0, stream>>>(
        d0o, d1W, d1b, d1_h, 1, BB, 50, 100, 0, 0, 0, 0, 1);
    bn_k<<<50, 256, 0, stream>>>(d1_h, d1g, d1be, d1o, BB, 50);
    small_lin_k<<<cdiv((long long)BB * HH, 256), 256, 0, stream>>>(
        d1o, d2W, d2b, d2_h, 1, BB, HH, 50, 0, 0, 0, 0, 1);
    bn_k<<<HH, 256, 0, stream>>>(d2_h, d2g, d2be, d2o, BB, HH);

    // ---- fusion heads ----
    concat2_k<<<cdiv((long long)BB * 2 * HH, 256), 256, 0, stream>>>(root_out, d2o, f1_in, HH, HH);
    small_lin_k<<<cdiv((long long)BB * HH, 256), 256, 0, stream>>>(
        f1_in, f1_W, f1_b, f1_h, 1, BB, HH, 2 * HH, 0, 0, 0, 0, 1);
    bn_k<<<HH, 256, 0, stream>>>(f1_h, f1_g, f1_be, f1o, BB, HH);
    concat2_k<<<cdiv((long long)BB * 2 * HH, 256), 256, 0, stream>>>(jjh_out, d2o, f2_in, HH, HH);
    small_lin_k<<<cdiv((long long)BB * HH, 256), 256, 0, stream>>>(
        f2_in, f2_W, f2_b, f2_h, 1, BB, HH, 2 * HH, 0, 0, 0, 0, 1);
    bn_k<<<HH, 256, 0, stream>>>(f2_h, f2_g, f2_be, f2o, BB, HH);

    aux_final_k<<<cdiv(BB, 256), 256, 0, stream>>>(
        f1o, f2o, f1aW, f1ab, f1oW, f1ob, f2aW, f2ab, f2oW, f2ob, (float*)d_out);
}
